// ParaTransformerLayer_446676599076
// MI455X (gfx1250) — compile-verified, matrix operand order checked by reading, not run on hardware
//
#include <hip/hip_runtime.h>
#include <math.h>

#define B_ 16
#define T_ 64
#define N_ 24
#define D_ 128
#define H_ 8
#define DEP_ 16
#define MAXREL_ 16
#define CTX_ 120
#define EPSILON_ 0.01f
#define TANH_SCALE_ 5.0f
#define LN_EPS_ 1e-5f
#define SCALE_ 0.25f   /* 1/sqrt(DEP) */

typedef __attribute__((ext_vector_type(16))) _Float16 v16h;
typedef __attribute__((ext_vector_type(8)))  _Float16 v8h;
typedef __attribute__((ext_vector_type(8)))  float    v8f;

// ---------------------------------------------------------------------------
// Weight convert+transpose: W[n][k][c] (f32) -> Wt[n][c][k] (f16).
// Writes coalesced along k.
// ---------------------------------------------------------------------------
__global__ void wt_convert_kernel(const float* __restrict__ W,
                                  _Float16* __restrict__ Wt, int total)
{
    int gid = blockIdx.x * blockDim.x + threadIdx.x;
    if (gid >= total) return;
    int k = gid & (D_ - 1);
    int c = (gid >> 7) & (D_ - 1);
    int n = gid >> 14;
    Wt[gid] = (_Float16)W[(n * D_ + k) * D_ + c];
}

// ---------------------------------------------------------------------------
// Generic WMMA GEMM:  Y[j, r, :] = X[j, r, :] @ W[j] + bias[j]   (K = NC = 128)
// One wave computes a 16x32 output strip (2 column tiles) via
// v_wmma_f32_16x16x32_f16, sharing the A fragment across both tiles.
// Weights are pre-transposed f16: Wt[j][col][k], so B loads are b128.
// ---------------------------------------------------------------------------
__global__ void gemm_wmma_kernel(const float* __restrict__ X,
                                 const _Float16* __restrict__ Wt,
                                 const float* __restrict__ Bv,
                                 float* __restrict__ Y,
                                 int row_tiles,       // per joint
                                 int x_joint_off, int x_row_stride,
                                 int y_joint_off, int y_row_stride,
                                 int w_joint_stride, int bias_joint_stride,
                                 int total_tiles, int relu)
{
    int wave = (blockIdx.x * blockDim.x + threadIdx.x) >> 5;
    if (wave >= total_tiles) return;          // uniform per wave: EXEC stays all-1s

    int lane = threadIdx.x & 31;
    int half = lane >> 4;                     // 0: lanes 0-15, 1: lanes 16-31
    int l16  = lane & 15;

    int per_joint = row_tiles << 2;           // row_tiles * 4 column-pairs
    int joint   = wave / per_joint;
    int rem     = wave - joint * per_joint;
    int rowTile = rem >> 2;
    int colPair = rem & 3;
    int col0 = colPair * 32 + l16;            // columns handled by this lane
    int col1 = col0 + 16;

    // A: lane holds row (rowTile*16 + l16); K pattern 0-7/16-23 (half 0) or 8-15/24-31 (half 1)
    const float* arow = X + joint * x_joint_off + (rowTile * 16 + l16) * x_row_stride;
    // B: lane holds column col0/col1; Wt row is contiguous in k.
    const _Float16* w0 = Wt + joint * w_joint_stride + col0 * D_;
    const _Float16* w1 = Wt + joint * w_joint_stride + col1 * D_;

    v8f acc0 = {}, acc1 = {};
#pragma unroll
    for (int k0 = 0; k0 < D_; k0 += 32) {
        v16h a;
#pragma unroll
        for (int i = 0; i < 16; ++i) {
            int kk = k0 + i + ((i >= 8) ? 8 : 0) + (half << 3);
            a[i] = (_Float16)arow[kk];
        }
        int kb = k0 + (half << 3);
        v8h b0lo = *(const v8h*)(w0 + kb);
        v8h b0hi = *(const v8h*)(w0 + kb + 16);
        v8h b1lo = *(const v8h*)(w1 + kb);
        v8h b1hi = *(const v8h*)(w1 + kb + 16);
        v16h b0, b1;
#pragma unroll
        for (int i = 0; i < 8; ++i) {
            b0[i] = b0lo[i]; b0[i + 8] = b0hi[i];
            b1[i] = b1lo[i]; b1[i + 8] = b1hi[i];
        }
        acc0 = __builtin_amdgcn_wmma_f32_16x16x32_f16(
            false, a, false, b0, (short)0, acc0, false, false);
        acc1 = __builtin_amdgcn_wmma_f32_16x16x32_f16(
            false, a, false, b1, (short)0, acc1, false, false);
    }

    float bc0 = Bv[joint * bias_joint_stride + col0];
    float bc1 = Bv[joint * bias_joint_stride + col1];
    float* ybase = Y + joint * y_joint_off;
#pragma unroll
    for (int m = 0; m < 8; ++m) {
        int row = rowTile * 16 + half * 8 + m;   // C/D layout: VGPR m = row, +8 for upper half
        float v0 = acc0[m] + bc0;
        float v1 = acc1[m] + bc1;
        if (relu) { v0 = fmaxf(v0, 0.0f); v1 = fmaxf(v1, 0.0f); }
        ybase[row * y_row_stride + col0] = v0;
        ybase[row * y_row_stride + col1] = v1;
    }
}

// ---------------------------------------------------------------------------
// Temporal attention (causal) with relative position bias, online softmax.
// One thread per (b, n, h, tq).
// ---------------------------------------------------------------------------
__global__ void temporal_attn_kernel(const float* __restrict__ q,
                                     const float* __restrict__ k,
                                     const float* __restrict__ v,
                                     const float* __restrict__ relk,
                                     const float* __restrict__ relv,
                                     float* __restrict__ out)
{
    int gid = blockIdx.x * blockDim.x + threadIdx.x;
    if (gid >= B_ * N_ * H_ * T_) return;
    int tq = gid % T_;
    int h  = (gid / T_) % H_;
    int n  = (gid / (T_ * H_)) % N_;
    int b  =  gid / (T_ * H_ * N_);

    const float* qp = q + ((size_t)((b * T_ + tq) * N_ + n)) * D_ + h * DEP_;
    float qv[DEP_];
#pragma unroll
    for (int d = 0; d < DEP_; ++d) qv[d] = qp[d];

    float mrun = -3.0e38f, lrun = 0.0f, acc[DEP_];
#pragma unroll
    for (int d = 0; d < DEP_; ++d) acc[d] = 0.0f;

    for (int tk = 0; tk <= tq; ++tk) {
        int rel = tk - tq;
        if (rel < -MAXREL_) rel = -MAXREL_;
        int idx = rel + MAXREL_;
        const float* kp = k + ((size_t)((b * T_ + tk) * N_ + n)) * D_ + h * DEP_;
        const float* rk = relk + idx * DEP_;
        float s = 0.0f;
#pragma unroll
        for (int d = 0; d < DEP_; ++d) s += qv[d] * (kp[d] + rk[d]);
        s *= SCALE_;
        float mn   = fmaxf(mrun, s);
        float corr = __expf(mrun - mn);
        float p    = __expf(s - mn);
        lrun = lrun * corr + p;
        const float* vp = v + ((size_t)((b * T_ + tk) * N_ + n)) * D_ + h * DEP_;
        const float* rv = relv + idx * DEP_;
#pragma unroll
        for (int d = 0; d < DEP_; ++d) acc[d] = acc[d] * corr + p * (vp[d] + rv[d]);
        mrun = mn;
    }
    float inv = 1.0f / lrun;
    float* op = out + ((size_t)((b * T_ + tq) * N_ + n)) * D_ + h * DEP_;
#pragma unroll
    for (int d = 0; d < DEP_; ++d) op[d] = acc[d] * inv;
}

// ---------------------------------------------------------------------------
// Spatial attention over joints. One thread per (b, t, h, n).
// ---------------------------------------------------------------------------
__global__ void spatial_attn_kernel(const float* __restrict__ q,
                                    const float* __restrict__ k,
                                    const float* __restrict__ v,
                                    float* __restrict__ out)
{
    int gid = blockIdx.x * blockDim.x + threadIdx.x;
    if (gid >= B_ * T_ * H_ * N_) return;
    int n = gid % N_;
    int h = (gid / N_) % H_;
    int t = (gid / (N_ * H_)) % T_;
    int b =  gid / (N_ * H_ * T_);

    size_t rowbase = (size_t)(b * T_ + t) * N_;
    const float* qp = q + (rowbase + n) * D_ + h * DEP_;
    float qv[DEP_];
#pragma unroll
    for (int d = 0; d < DEP_; ++d) qv[d] = qp[d];

    float sc[N_];
    float mx = -3.0e38f;
    for (int m = 0; m < N_; ++m) {
        const float* kp = k + (rowbase + m) * D_ + h * DEP_;
        float s = 0.0f;
#pragma unroll
        for (int d = 0; d < DEP_; ++d) s += qv[d] * kp[d];
        s *= SCALE_;
        sc[m] = s;
        mx = fmaxf(mx, s);
    }
    float lsum = 0.0f;
    for (int m = 0; m < N_; ++m) { sc[m] = __expf(sc[m] - mx); lsum += sc[m]; }
    float inv = 1.0f / lsum;

    float acc[DEP_];
#pragma unroll
    for (int d = 0; d < DEP_; ++d) acc[d] = 0.0f;
    for (int m = 0; m < N_; ++m) {
        const float* vp = v + (rowbase + m) * D_ + h * DEP_;
        float w = sc[m] * inv;
#pragma unroll
        for (int d = 0; d < DEP_; ++d) acc[d] += w * vp[d];
    }
    float* op = out + (rowbase + n) * D_ + h * DEP_;
#pragma unroll
    for (int d = 0; d < DEP_; ++d) op[d] = acc[d];
}

// ---------------------------------------------------------------------------
// Cross attention vs. context (CTX=120), online softmax. Thread per (b,n,h,t).
// ---------------------------------------------------------------------------
__global__ void cross_attn_kernel(const float* __restrict__ cq,
                                  const float* __restrict__ ck,
                                  const float* __restrict__ cv,
                                  float* __restrict__ out)
{
    int gid = blockIdx.x * blockDim.x + threadIdx.x;
    if (gid >= B_ * N_ * H_ * T_) return;
    int t = gid % T_;
    int h = (gid / T_) % H_;
    int n = (gid / (T_ * H_)) % N_;
    int b =  gid / (T_ * H_ * N_);

    const float* qp = cq + ((size_t)((b * T_ + t) * N_ + n)) * D_ + h * DEP_;
    float qv[DEP_];
#pragma unroll
    for (int d = 0; d < DEP_; ++d) qv[d] = qp[d];

    float mrun = -3.0e38f, lrun = 0.0f, acc[DEP_];
#pragma unroll
    for (int d = 0; d < DEP_; ++d) acc[d] = 0.0f;

    for (int c = 0; c < CTX_; ++c) {
        const float* kp = ck + ((size_t)(b * CTX_ + c)) * D_ + h * DEP_;
        float s = 0.0f;
#pragma unroll
        for (int d = 0; d < DEP_; ++d) s += qv[d] * kp[d];
        s *= SCALE_;
        float mn   = fmaxf(mrun, s);
        float corr = __expf(mrun - mn);
        float p    = __expf(s - mn);
        lrun = lrun * corr + p;
        const float* vp = cv + ((size_t)(b * CTX_ + c)) * D_ + h * DEP_;
#pragma unroll
        for (int d = 0; d < DEP_; ++d) acc[d] = acc[d] * corr + p * vp[d];
        mrun = mn;
    }
    float inv = 1.0f / lrun;
    float* op = out + ((size_t)((b * T_ + t) * N_ + n)) * D_ + h * DEP_;
#pragma unroll
    for (int d = 0; d < DEP_; ++d) op[d] = acc[d] * inv;
}

// ---------------------------------------------------------------------------
// out = LayerNorm(a + b? + c?) over D=128. One wave32 per row, 4 elems/lane.
// ---------------------------------------------------------------------------
__global__ void add_ln_kernel(const float* __restrict__ a,
                              const float* __restrict__ b,
                              const float* __restrict__ c,
                              const float* __restrict__ g,
                              const float* __restrict__ be,
                              float* __restrict__ out, int rows)
{
    int wave = (blockIdx.x * blockDim.x + threadIdx.x) >> 5;
    int lane = threadIdx.x & 31;
    if (wave >= rows) return;
    size_t base = (size_t)wave * D_;

    float x[4], s = 0.0f, ss = 0.0f;
#pragma unroll
    for (int j = 0; j < 4; ++j) {
        int idx = lane + j * 32;
        float t = a[base + idx];
        if (b) t += b[base + idx];
        if (c) t += c[base + idx];
        x[j] = t; s += t; ss += t * t;
    }
#pragma unroll
    for (int o = 16; o >= 1; o >>= 1) {
        s  += __shfl_xor(s,  o, 32);
        ss += __shfl_xor(ss, o, 32);
    }
    float mean = s * (1.0f / D_);
    float var  = ss * (1.0f / D_) - mean * mean;    // biased (matches jnp.var)
    float r    = rsqrtf(var + LN_EPS_);
#pragma unroll
    for (int j = 0; j < 4; ++j) {
        int idx = lane + j * 32;
        out[base + idx] = (x[j] - mean) * r * g[idx] + be[idx];
    }
}

// ---------------------------------------------------------------------------
// LayerNorm over DEP=16 of kv (B,CTX,H,DEP), in-place. Thread per (b,c,h).
// ---------------------------------------------------------------------------
__global__ void kv_ln_kernel(float* __restrict__ kv,
                             const float* __restrict__ g,
                             const float* __restrict__ be)
{
    int gid = blockIdx.x * blockDim.x + threadIdx.x;
    if (gid >= B_ * CTX_ * H_) return;
    float* p = kv + (size_t)gid * DEP_;
    float s = 0.0f, ss = 0.0f;
#pragma unroll
    for (int d = 0; d < DEP_; ++d) { float t = p[d]; s += t; ss += t * t; }
    float mean = s * (1.0f / DEP_);
    float var  = ss * (1.0f / DEP_) - mean * mean;
    float r    = rsqrtf(var + LN_EPS_);
#pragma unroll
    for (int d = 0; d < DEP_; ++d) p[d] = (p[d] - mean) * r * g[d] + be[d];
}

// ---------------------------------------------------------------------------
// PosteriorRotation: skew = 5*tanh((rq . kv_c)/5), A skew-symmetric from
// tril_indices(16,-1); rot = rep + eps * A^T-apply. Thread per (b,t,n,h).
// ---------------------------------------------------------------------------
__global__ void rotation_kernel(const float* __restrict__ rep_buf,
                                const float* __restrict__ rq_buf,
                                const float* __restrict__ kv,
                                float* __restrict__ out)
{
    int gid = blockIdx.x * blockDim.x + threadIdx.x;
    if (gid >= B_ * T_ * N_ * H_) return;
    int h = gid % H_;
    int n = (gid / H_) % N_;
    int t = (gid / (H_ * N_)) % T_;
    int b =  gid / (H_ * N_ * T_);

    size_t off = ((size_t)((b * T_ + t) * N_ + n)) * D_ + h * DEP_;
    float rep[DEP_], rq[DEP_], rot[DEP_];
#pragma unroll
    for (int d = 0; d < DEP_; ++d) { rep[d] = rep_buf[off + d]; rq[d] = rq_buf[off + d]; rot[d] = rep[d]; }

    int r = 1, cc = 0;   // np.tril_indices(16,-1) order: (1,0),(2,0),(2,1),(3,0)...
    for (int c = 0; c < CTX_; ++c) {
        const float* kp = kv + ((size_t)(b * CTX_ + c)) * D_ + h * DEP_;
        float s = 0.0f;
#pragma unroll
        for (int d = 0; d < DEP_; ++d) s += rq[d] * kp[d];
        s = TANH_SCALE_ * tanhf(s * (1.0f / TANH_SCALE_));
        rot[r]  += EPSILON_ * s * rep[cc];
        rot[cc] -= EPSILON_ * s * rep[r];
        if (++cc == r) { ++r; cc = 0; }
    }
#pragma unroll
    for (int d = 0; d < DEP_; ++d) out[off + d] = rot[d];
}

// ---------------------------------------------------------------------------
static void launch_gemm(const float* X, const _Float16* Wt, const float* Bv, float* Y,
                        int n_joints, int rows_per_joint,
                        int x_joint_off, int x_row_stride,
                        int y_joint_off, int y_row_stride,
                        int w_joint_stride, int bias_joint_stride,
                        int relu, hipStream_t stream)
{
    int row_tiles = rows_per_joint / 16;
    int total = n_joints * row_tiles * 4;         // 4 column-pairs of 32
    int blocks = (total + 3) / 4;                 // 4 waves / block of 128
    gemm_wmma_kernel<<<blocks, 128, 0, stream>>>(
        X, Wt, Bv, Y, row_tiles, x_joint_off, x_row_stride,
        y_joint_off, y_row_stride, w_joint_stride, bias_joint_stride, total, relu);
}

static void launch_wt(const float* W, _Float16* Wt, int joints, hipStream_t stream)
{
    int total = joints * D_ * D_;
    wt_convert_kernel<<<(total + 255) / 256, 256, 0, stream>>>(W, Wt, total);
}

extern "C" void kernel_launch(void* const* d_in, const int* in_sizes, int n_in,
                              void* d_out, int out_size, void* d_ws, size_t ws_size,
                              hipStream_t stream)
{
    const float* x    = (const float*)d_in[0];
    const float* ctx  = (const float*)d_in[1];
    const float* t_wq = (const float*)d_in[2];  const float* t_bq = (const float*)d_in[3];
    const float* t_wk = (const float*)d_in[4];  const float* t_bk = (const float*)d_in[5];
    const float* t_wv = (const float*)d_in[6];  const float* t_bv = (const float*)d_in[7];
    const float* t_wo = (const float*)d_in[8];  const float* t_bo = (const float*)d_in[9];
    const float* relk = (const float*)d_in[10]; const float* relv = (const float*)d_in[11];
    const float* s_wq = (const float*)d_in[12]; const float* s_bq = (const float*)d_in[13];
    const float* s_wk = (const float*)d_in[14]; const float* s_bk = (const float*)d_in[15];
    const float* s_wv = (const float*)d_in[16]; const float* s_bv = (const float*)d_in[17];
    const float* s_wo = (const float*)d_in[18]; const float* s_bo = (const float*)d_in[19];
    const float* c_wq = (const float*)d_in[20]; const float* c_bq = (const float*)d_in[21];
    const float* c_wk = (const float*)d_in[22]; const float* c_bk = (const float*)d_in[23];
    const float* c_wv = (const float*)d_in[24]; const float* c_bv = (const float*)d_in[25];
    const float* c_wo = (const float*)d_in[26]; const float* c_bo = (const float*)d_in[27];
    const float* r_wq = (const float*)d_in[28]; const float* r_bq = (const float*)d_in[29];
    const float* r_wk = (const float*)d_in[30]; const float* r_bk = (const float*)d_in[31];
    const float* rln_g = (const float*)d_in[32]; const float* rln_b = (const float*)d_in[33];
    const float* f_w1 = (const float*)d_in[34]; const float* f_b1 = (const float*)d_in[35];
    const float* f_w2 = (const float*)d_in[36]; const float* f_b2 = (const float*)d_in[37];
    const float* ln1_g = (const float*)d_in[38]; const float* ln1_b = (const float*)d_in[39];
    const float* ln2_g = (const float*)d_in[40]; const float* ln2_b = (const float*)d_in[41];
    const float* ln3_g = (const float*)d_in[42]; const float* ln3_b = (const float*)d_in[43];

    const int SZ  = B_ * T_ * N_ * D_;       // 3,145,728 floats
    const int CSZ = B_ * CTX_ * D_;          // 245,760 floats
    float* ws   = (float*)d_ws;
    float* buf0 = ws;
    float* buf1 = ws + (size_t)SZ;
    float* buf2 = ws + (size_t)2 * SZ;
    float* buf3 = ws + (size_t)3 * SZ;
    float* buf4 = ws + (size_t)4 * SZ;
    float* buf5 = ws + (size_t)5 * SZ;
    float* cbuf0 = ws + (size_t)6 * SZ;
    float* cbuf1 = cbuf0 + CSZ;

    // f16 transposed-weight region (256B-aligned offsets)
    const int WPJ = N_ * D_ * D_;            // 393,216 f16 per per-joint tensor
    const int WSH = D_ * D_;                 // 16,384 f16 per shared tensor
    _Float16* wt = (_Float16*)(cbuf1 + CSZ);
    _Float16* wt_tq = wt;                    // per-joint: t_wq,t_wk,t_wv,t_wo,c_wq,r_wq,f_w1,f_w2
    _Float16* wt_tk = wt_tq + WPJ;
    _Float16* wt_tv = wt_tk + WPJ;
    _Float16* wt_to = wt_tv + WPJ;
    _Float16* wt_cq = wt_to + WPJ;
    _Float16* wt_rq = wt_cq + WPJ;
    _Float16* wt_f1 = wt_rq + WPJ;
    _Float16* wt_f2 = wt_f1 + WPJ;
    _Float16* wt_sq = wt_f2 + WPJ;           // shared: s_wq,s_wk,s_wv,s_wo,c_wk,c_wv,c_wo,r_wk
    _Float16* wt_sk = wt_sq + WSH;
    _Float16* wt_sv = wt_sk + WSH;
    _Float16* wt_so = wt_sv + WSH;
    _Float16* wt_ck = wt_so + WSH;
    _Float16* wt_cv = wt_ck + WSH;
    _Float16* wt_co = wt_cv + WSH;
    _Float16* wt_rk = wt_co + WSH;

    // ---- one-time (per call) weight convert + transpose to f16 ----
    launch_wt(t_wq, wt_tq, N_, stream);  launch_wt(t_wk, wt_tk, N_, stream);
    launch_wt(t_wv, wt_tv, N_, stream);  launch_wt(t_wo, wt_to, N_, stream);
    launch_wt(c_wq, wt_cq, N_, stream);  launch_wt(r_wq, wt_rq, N_, stream);
    launch_wt(f_w1, wt_f1, N_, stream);  launch_wt(f_w2, wt_f2, N_, stream);
    launch_wt(s_wq, wt_sq, 1, stream);   launch_wt(s_wk, wt_sk, 1, stream);
    launch_wt(s_wv, wt_sv, 1, stream);   launch_wt(s_wo, wt_so, 1, stream);
    launch_wt(c_wk, wt_ck, 1, stream);   launch_wt(c_wv, wt_cv, 1, stream);
    launch_wt(c_wo, wt_co, 1, stream);   launch_wt(r_wk, wt_rk, 1, stream);

    const int BT  = B_ * T_;                 // rows per joint (per-joint GEMM)
    const int BTN = B_ * T_ * N_;            // rows for shared GEMM
    const int BC  = B_ * CTX_;               // rows for context GEMM
    const int ROWS_LN = BTN;

    int thr_bnht = B_ * N_ * H_ * T_;        // 196,608
    int blk256_attn = (thr_bnht + 255) / 256;
    int blk256_ln   = ((ROWS_LN * 32) + 255) / 256;
    int blk256_kvln = (B_ * CTX_ * H_ + 255) / 256;

    // ---- temporal attention ----
    launch_gemm(x, wt_tq, t_bq, buf0, N_, BT, D_, N_ * D_, D_, N_ * D_, D_ * D_, D_, 0, stream);
    launch_gemm(x, wt_tk, t_bk, buf1, N_, BT, D_, N_ * D_, D_, N_ * D_, D_ * D_, D_, 0, stream);
    launch_gemm(x, wt_tv, t_bv, buf2, N_, BT, D_, N_ * D_, D_, N_ * D_, D_ * D_, D_, 0, stream);
    temporal_attn_kernel<<<blk256_attn, 256, 0, stream>>>(buf0, buf1, buf2, relk, relv, buf3);
    launch_gemm(buf3, wt_to, t_bo, buf4, N_, BT, D_, N_ * D_, D_, N_ * D_, D_ * D_, D_, 0, stream);

    // ---- spatial attention ----
    launch_gemm(x, wt_sq, s_bq, buf0, 1, BTN, 0, D_, 0, D_, 0, 0, 0, stream);
    launch_gemm(x, wt_sk, s_bk, buf1, 1, BTN, 0, D_, 0, D_, 0, 0, 0, stream);
    launch_gemm(x, wt_sv, s_bv, buf2, 1, BTN, 0, D_, 0, D_, 0, 0, 0, stream);
    spatial_attn_kernel<<<blk256_attn, 256, 0, stream>>>(buf0, buf1, buf2, buf3);
    launch_gemm(buf3, wt_so, s_bo, buf5, 1, BTN, 0, D_, 0, D_, 0, 0, 0, stream);

    // ---- out1 = LN(x + at + asp) ----
    add_ln_kernel<<<blk256_ln, 256, 0, stream>>>(x, buf4, buf5, ln1_g, ln1_b, buf0, ROWS_LN);

    // ---- cross attention ----
    launch_gemm(buf0, wt_cq, c_bq, buf1, N_, BT, D_, N_ * D_, D_, N_ * D_, D_ * D_, D_, 0, stream);
    launch_gemm(ctx, wt_ck, c_bk, cbuf0, 1, BC, 0, D_, 0, D_, 0, 0, 0, stream);
    launch_gemm(ctx, wt_cv, c_bv, cbuf1, 1, BC, 0, D_, 0, D_, 0, 0, 0, stream);
    cross_attn_kernel<<<blk256_attn, 256, 0, stream>>>(buf1, cbuf0, cbuf1, buf2);
    launch_gemm(buf2, wt_co, c_bo, buf3, 1, BTN, 0, D_, 0, D_, 0, 0, 0, stream);

    // ---- out2 = LN(out1 + co) ----
    add_ln_kernel<<<blk256_ln, 256, 0, stream>>>(buf0, buf3, nullptr, ln2_g, ln2_b, buf1, ROWS_LN);

    // ---- posterior rotation ----
    launch_gemm(ctx, wt_rk, r_bk, cbuf0, 1, BC, 0, D_, 0, D_, 0, 0, 0, stream);
    kv_ln_kernel<<<blk256_kvln, 256, 0, stream>>>(cbuf0, rln_g, rln_b);
    launch_gemm(buf1, wt_rq, r_bq, buf2, N_, BT, D_, N_ * D_, D_, N_ * D_, D_ * D_, D_, 0, stream);
    rotation_kernel<<<blk256_attn, 256, 0, stream>>>(buf1, buf2, cbuf0, buf3);

    // ---- per-joint FFN ----
    launch_gemm(buf3, wt_f1, f_b1, buf0, N_, BT, D_, N_ * D_, D_, N_ * D_, D_ * D_, D_, 1, stream);
    launch_gemm(buf0, wt_f2, f_b2, buf2, N_, BT, D_, N_ * D_, D_, N_ * D_, D_ * D_, D_, 0, stream);

    // ---- out = LN(rot + f) ----
    add_ln_kernel<<<blk256_ln, 256, 0, stream>>>(buf3, buf2, nullptr, ln3_g, ln3_b, (float*)d_out, ROWS_LN);
}
